// SeederCBOX_32736240730422
// MI455X (gfx1250) — matrix operand order repaired and checked by reading, not compiled, not run on hardware
//
#include <hip/hip_runtime.h>
#include <hip/hip_bf16.h>
#include <stdint.h>

typedef __attribute__((ext_vector_type(16))) _Float16 v16h;
typedef __attribute__((ext_vector_type(8)))  _Float16 v8h;
typedef __attribute__((ext_vector_type(8)))  float    v8f;

#define BATCH   128
#define HH      512
#define WW      512
#define HW      (HH*WW)
#define NSEEDS  1000
#define ER_R    5          // (11-1)/2 erosion radius
#define IGNORE_V (-255)

// ---------------- workspace layout (bytes) ----------------
#define OFF_H256      ((size_t)0)                         // 128*256*4   = 131072
#define OFF_H4096     ((size_t)131072)                    // 128*4096*4  = 2097152
#define OFF_TH        ((size_t)2228224)                   // 128 floats
#define OFF_TBIN      ((size_t)2228736)                   // 128 ints
#define OFF_CNTBG     ((size_t)2229248)                   // 128 ints
#define OFF_CNTFG     ((size_t)2229760)                   // 128 ints
#define OFF_PARAM_END ((size_t)2230272)
#define OFF_BUF1      OFF_PARAM_END                       // f16 row-sums: 128*HW*2 = 64 MB
#define OFF_MFG       (OFF_BUF1 + (size_t)BATCH*HW*2)     // u8 fg mask: 32 MB
#define OFF_MBG       (OFF_MFG  + (size_t)BATCH*HW)       // u8 bg mask: 32 MB

// ---------------- hashing (deterministic, stateless) ----------------
__device__ __forceinline__ unsigned h32(unsigned x) {
  x ^= x >> 16; x *= 0x7feb352dU;
  x ^= x >> 15; x *= 0x846ca68bU;
  x ^= x >> 16; return x;
}
__device__ __forceinline__ unsigned hmix(unsigned a, unsigned b, unsigned c) {
  return h32(a * 0x9E3779B1U ^ b * 0x85EBCA77U ^ c * 0xC2B2AE3DU);
}

// ---------------- 0: clear params/histograms ----------------
__global__ void clear_kernel(int* p, int n) {
  int i = blockIdx.x * blockDim.x + threadIdx.x;
  if (i < n) p[i] = 0;
}

// ---------------- 1: per-image histograms in LDS ----------------
__global__ void hist_kernel(const float* __restrict__ x,
                            int* __restrict__ h256, int* __restrict__ h4096) {
  __shared__ int s256[256];
  __shared__ int s4096[4096];
  const int tid = threadIdx.x;
  const int img = blockIdx.y;
  s256[tid] = 0;
  for (int i = tid; i < 4096; i += 256) s4096[i] = 0;
  __syncthreads();
  const float* im = x + (size_t)img * HW;
  const int base = blockIdx.x * 8192;          // 32 blocks * 8192 px = HW
  for (int i = 0; i < 32; ++i) {
    const float v = im[base + i * 256 + tid];
    int c255 = (int)floorf(v * 255.f);
    c255 = c255 < 0 ? 0 : (c255 > 255 ? 255 : c255);
    atomicAdd(&s256[c255], 1);
    int b = (int)(v * 4096.f);
    b = b < 0 ? 0 : (b > 4095 ? 4095 : b);
    atomicAdd(&s4096[b], 1);
  }
  __syncthreads();
  atomicAdd(&h256[img * 256 + tid], s256[tid]);
  for (int i = tid; i < 4096; i += 256) atomicAdd(&h4096[img * 4096 + i], s4096[i]);
}

// ---------------- 2: Otsu + background quantile (tiny serial scans) ----------------
__global__ void otsu_kernel(const int* __restrict__ h256, const int* __restrict__ h4096,
                            float* __restrict__ th, int* __restrict__ tbin,
                            int* __restrict__ cntbg) {
  const int img = blockIdx.x;
  if (threadIdx.x != 0) return;
  const int* h = h256 + img * 256;
  const float total = (float)HW;
  float mT = 0.f; int nz = 0;
  for (int i = 0; i < 256; ++i) { mT += (float)h[i] * (float)i; if (h[i] > 0) ++nz; }
  float w = 0.f, m = 0.f, best = -1.f; int bt = 0;
  for (int t = 0; t < 256; ++t) {
    w += (float)h[t]; m += (float)h[t] * (float)t;
    const float denom = w * (total - w);
    float s = 0.f;
    if (denom > 0.f) { const float d = mT * w - total * m; s = d * d / (denom * total * total); }
    if (s > best) { best = s; bt = t; }      // first max, like jnp.argmax
  }
  float thr = (float)bt;
  const int* h4 = h4096 + img * 4096;
  if (nz <= 1) {                             // degenerate -> median fallback
    long long cum = 0;
    for (int i = 0; i < 4096; ++i) { cum += h4[i]; if (cum * 2 >= HW) { thr = ((i + 0.5f) / 4096.f) * 255.f; break; } }
  }
  if (thr == 0.f) thr = 1.f; else if (thr == 255.f) thr = 254.f;
  th[img] = thr;
  // nbr_bg = ceil(z*HW), z ~ U(0.3,0.7) (hash-derived)
  const float z = 0.3f + 0.4f * ((float)h32(0xB5297A4DU ^ (unsigned)img) * (1.0f / 4294967296.0f));
  const long long nbr = (long long)ceilf(z * (float)HW);
  long long cum = 0; int t4 = 4095;
  for (int i = 0; i < 4096; ++i) { cum += h4[i]; if (cum >= nbr) { t4 = i; break; } }
  tbin[img] = t4;
  cntbg[img] = (int)cum;                     // #eligible background pixels
}

// ---------------- 3: row-direction zero-count via banded-ones WMMA ----------------
// RowSum[r,c] = sum_{|k-c|<=5, in-bounds} Z[r,k],  Z = 1 - (cam255 > th)
// Block = 8 waves covering rows [r0,r0+16) x cols [cb0,cb0+128).
// Z window (16 x 176 f16, halo included) is staged in LDS with branchless,
// coalesced global loads; A fragments are two aligned ds_load_b128 per K-step.
#define RS_COLS 176
__global__ void rowsum_wmma_kernel(const float* __restrict__ x, const float* __restrict__ th,
                                   _Float16* __restrict__ buf1) {
  __shared__ __align__(16) _Float16 zt[16 * RS_COLS];
  const int img = blockIdx.y;
  const int r0  = (blockIdx.x >> 2) * 16;
  const int cb0 = (blockIdx.x & 3) * 128;
  const float t = th[img];
  const float* im = x + (size_t)img * HW;

  // stage Z: cols [cb0-16, cb0+160), zero-filled outside the image (branchless)
  for (int i = threadIdx.x; i < 16 * RS_COLS; i += 256) {
    const int row = i / RS_COLS;
    const int co  = i % RS_COLS;
    const int col = cb0 - 16 + co;
    const int ccl = col < 0 ? 0 : (col > (WW - 1) ? (WW - 1) : col);
    const float v = im[(size_t)(r0 + row) * WW + ccl];
    const int valid = (col >= 0) & (col < WW);
    const float zf = (valid && !(floorf(v * 255.f) > t)) ? 1.f : 0.f;
    zt[row * RS_COLS + co] = (_Float16)zf;
  }
  __syncthreads();

  const int wv   = threadIdx.x >> 5;          // wave -> column tile
  const int lane = threadIdx.x & 31;
  const int half = lane >> 4, mn = lane & 15;
  const int c0 = cb0 + wv * 16;
  v8f acc = {};
#pragma unroll
  for (int kt = 0; kt < 2; ++kt) {
    // A fragment (16x32 f16, ISA 7.12.2): per lane two contiguous 8-elem runs
    const int off = wv * 16 + kt * 32 + half * 8;   // multiple of 8 -> 16B aligned
    const v8h ra = *(const v8h*)&zt[mn * RS_COLS + off];
    const v8h rb = *(const v8h*)&zt[mn * RS_COLS + off + 16];
    v16h a, b;
#pragma unroll
    for (int i = 0; i < 8; ++i) { a[i] = ra[i]; a[i + 8] = rb[i]; }
    // B fragment (32x16): banded-ones, K = half*8 + e + (e>>3)*8, N = lane&15 (pure ALU)
    const int kb = c0 - 16 + kt * 32;
#pragma unroll
    for (int e = 0; e < 16; ++e) {
      const int d = (kb + (half << 3) + e + ((e >> 3) << 3)) - (c0 + mn);
      b[e] = (_Float16)((d >= -ER_R && d <= ER_R) ? 1.f : 0.f);
    }
    acc = __builtin_amdgcn_wmma_f32_16x16x32_f16(false, a, false, b, (short)0, acc, false, false);
  }
  _Float16* ob = buf1 + (size_t)img * HW;
#pragma unroll
  for (int g = 0; g < 8; ++g)                // D layout: M = half*8 + g, N = lane&15
    ob[(size_t)(r0 + half * 8 + g) * WW + (c0 + mn)] = (_Float16)acc[g];
}

// ---------------- 4: col-direction via WMMA -> eroded mask + fg count ----------------
// Stage rows [r0-16, r0+48) x cols [cb0,cb0+128) of buf1 into LDS with coalesced
// global_load_b128; fragment B reads are strided ds_load_u16.
#define CS_STRIDE 136   // 128 data cols + pad, keeps rows 16B-aligned
__global__ void colsum_wmma_kernel(const _Float16* __restrict__ buf1,
                                   unsigned char* __restrict__ mfg, int* __restrict__ cntfg) {
  __shared__ __align__(16) _Float16 st[64 * CS_STRIDE];
  __shared__ int scnt[256];
  const int img = blockIdx.y;
  const int r0  = (blockIdx.x >> 2) * 16;
  const int cb0 = (blockIdx.x & 3) * 128;
  const _Float16* ib = buf1 + (size_t)img * HW;

  // stage: 64 rows x 128 cols = 1024 runs of 8 f16 (16 bytes each)
  for (int i = 0; i < 4; ++i) {
    const int run = i * 256 + threadIdx.x;
    const int row = run >> 4;                 // 0..63
    const int cw  = (run & 15) * 8;           // 0..120
    const int rr  = r0 - 16 + row;
    v8h v = {};
    if (rr >= 0 && rr < HH) v = *(const v8h*)&ib[(size_t)rr * WW + cb0 + cw];
    *(v8h*)&st[row * CS_STRIDE + cw] = v;
  }
  __syncthreads();

  const int wv   = threadIdx.x >> 5;
  const int lane = threadIdx.x & 31;
  const int half = lane >> 4, mn = lane & 15;
  const int c0 = cb0 + wv * 16;
  v8f acc = {};
#pragma unroll
  for (int kt = 0; kt < 2; ++kt) {
    v16h a, b;
#pragma unroll
    for (int e = 0; e < 16; ++e) {
      // A: banded-ones W[r,k] = |r-k|<=5 (pure ALU)
      const int v = e >> 1, lo = e & 1;
      const int kl = ((v < 4) ? 0 : 16) + (half << 3) + ((v & 3) << 1) + lo;
      const int kb = r0 - 16 + kt * 32;
      const int d = (kb + kl) - (r0 + mn);
      a[e] = (_Float16)((d >= -ER_R && d <= ER_R) ? 1.f : 0.f);
      // B: staged row-sum tile; LDS row = kt*32 + half*8 + e + (e>>3)*8 (always in [0,64))
      b[e] = st[(kt * 32 + (half << 3) + e + ((e >> 3) << 3)) * CS_STRIDE + wv * 16 + mn];
    }
    acc = __builtin_amdgcn_wmma_f32_16x16x32_f16(false, a, false, b, (short)0, acc, false, false);
  }
  unsigned char* om = mfg + (size_t)img * HW;
  int local = 0;
#pragma unroll
  for (int g = 0; g < 8; ++g) {
    const int er = (acc[g] < 0.5f) ? 1 : 0;  // zero zeros in 11x11 window -> eroded ROI
    om[(size_t)(r0 + half * 8 + g) * WW + (c0 + mn)] = (unsigned char)er;
    local += er;
  }
  scnt[threadIdx.x] = local;
  __syncthreads();
  for (int s = 128; s > 0; s >>= 1) {
    if (threadIdx.x < s) scnt[threadIdx.x] += scnt[threadIdx.x + s];
    __syncthreads();
  }
  if (threadIdx.x == 0) atomicAdd(&cntfg[img], scnt[0]);
}

// ---------------- 5: hash-based uniform seed selection ----------------
__global__ void sample_kernel(const float* __restrict__ x, const int* __restrict__ tbin,
                              const int* __restrict__ cntbg, const int* __restrict__ cntfg,
                              unsigned char* __restrict__ mfg, unsigned char* __restrict__ mbg) {
  const size_t idx = (size_t)blockIdx.x * blockDim.x + threadIdx.x;
  if (idx >= (size_t)BATCH * HW) return;
  const int img = (int)(idx / HW);
  const unsigned p = (unsigned)(idx % HW);
  // foreground: select with prob min(1, 1000/cntfg) among eroded-ROI pixels
  bool fg = false;
  if (mfg[idx]) {
    const unsigned cF = (unsigned)(cntfg[img] > 0 ? cntfg[img] : 1);
    const unsigned hv = hmix((unsigned)img, p, 0x01234567U);
    fg = ((unsigned long long)hv * cF) < ((unsigned long long)NSEEDS << 32);
  }
  mfg[idx] = fg ? 1 : 0;
  // background: eligible = among nbr_bg lowest activations (rank-select bin threshold)
  bool bg = false;
  const float v = x[idx];
  int bin = (int)(v * 4096.f);
  bin = bin < 0 ? 0 : (bin > 4095 ? 4095 : bin);
  if (bin <= tbin[img]) {
    const unsigned cB = (unsigned)(cntbg[img] > 0 ? cntbg[img] : 1);
    const unsigned hv = hmix((unsigned)img, p, 0x89ABCDEFU);
    bg = ((unsigned long long)hv * cB) < ((unsigned long long)NSEEDS << 32);
  }
  mbg[idx] = bg ? 1 : 0;
}

// ---------------- 6: fused 3x3 dilation + mutual exclusion + labels ----------------
__global__ void finalize_kernel(const unsigned char* __restrict__ mfg,
                                const unsigned char* __restrict__ mbg,
                                int* __restrict__ out) {
  const size_t idx = (size_t)blockIdx.x * blockDim.x + threadIdx.x;
  if (idx >= (size_t)BATCH * HW) return;
  const int img = (int)(idx / HW);
  const int p = (int)(idx % HW);
  const int r = p / WW, c = p % WW;
  const unsigned char* f = mfg + (size_t)img * HW;
  const unsigned char* b = mbg + (size_t)img * HW;
  int dfg = 0, dbg = 0;
  for (int dr = -1; dr <= 1; ++dr) {
    const int rr = r + dr;
    if (rr < 0 || rr >= HH) continue;
    for (int dc = -1; dc <= 1; ++dc) {
      const int cc = c + dc;
      if (cc < 0 || cc >= WW) continue;
      const size_t q = (size_t)rr * WW + cc;
      dfg |= f[q]; dbg |= b[q];
    }
  }
  int o = IGNORE_V;
  if (dfg && !dbg) o = 1;
  else if (dbg && !dfg) o = 0;               // overlap -> both zeroed -> IGNORE
  out[idx] = o;
}

// ---------------- launcher ----------------
extern "C" void kernel_launch(void* const* d_in, const int* in_sizes, int n_in,
                              void* d_out, int out_size, void* d_ws, size_t ws_size,
                              hipStream_t stream) {
  (void)in_sizes; (void)n_in; (void)out_size; (void)ws_size;
  const float* x = (const float*)d_in[0];          // (128,1,512,512) f32
  int* out = (int*)d_out;                          // (128,512,512) i32 labels
  char* ws = (char*)d_ws;                          // needs ~131 MB scratch
  int*      h256  = (int*)(ws + OFF_H256);
  int*      h4096 = (int*)(ws + OFF_H4096);
  float*    th    = (float*)(ws + OFF_TH);
  int*      tbin  = (int*)(ws + OFF_TBIN);
  int*      cntbg = (int*)(ws + OFF_CNTBG);
  int*      cntfg = (int*)(ws + OFF_CNTFG);
  _Float16* buf1  = (_Float16*)(ws + OFF_BUF1);
  unsigned char* mfg = (unsigned char*)(ws + OFF_MFG);
  unsigned char* mbg = (unsigned char*)(ws + OFF_MBG);

  const int clear_n = (int)(OFF_PARAM_END / 4);
  clear_kernel<<<(clear_n + 255) / 256, 256, 0, stream>>>((int*)ws, clear_n);
  hist_kernel<<<dim3(32, BATCH), 256, 0, stream>>>(x, h256, h4096);
  otsu_kernel<<<BATCH, 32, 0, stream>>>(h256, h4096, th, tbin, cntbg);
  rowsum_wmma_kernel<<<dim3(128, BATCH), 256, 0, stream>>>(x, th, buf1);
  colsum_wmma_kernel<<<dim3(128, BATCH), 256, 0, stream>>>(buf1, mfg, cntfg);
  const size_t npix = (size_t)BATCH * HW;
  const int pblocks = (int)((npix + 255) / 256);
  sample_kernel<<<pblocks, 256, 0, stream>>>(x, tbin, cntbg, cntfg, mfg, mbg);
  finalize_kernel<<<pblocks, 256, 0, stream>>>(mfg, mbg, out);
}
